// HyperbolicAttentionStable_44727789420748
// MI455X (gfx1250) — compile-verified, matrix operand order checked by reading, not run on hardware
//
#include <hip/hip_runtime.h>
#include <hip/hip_bf16.h>
#include <math.h>

typedef __bf16 bf16_t;
typedef __attribute__((ext_vector_type(16))) __bf16        v16bf;
typedef __attribute__((ext_vector_type(8)))  float         v8f;
typedef __attribute__((ext_vector_type(4)))  unsigned int  u32x4;

union Frag16 { v16bf v; u32x4 q[2]; };

// Load a 16x32 bf16 WMMA A/B fragment from row-major memory (ld = elements per row).
// Per ISA 16-bit A-matrix layout: lanes 0-15 hold K {0..7,16..23}, lanes 16-31 hold K {8..15,24..31},
// row = lane%16. Two 16-byte loads per lane.
__device__ inline v16bf load_frag_b16(const bf16_t* __restrict__ base, int ld) {
  int lane = threadIdx.x & 31;
  int row  = lane & 15;
  int kb   = (lane >> 4) << 3;          // 0 or 8 elements
  const bf16_t* p = base + (size_t)row * ld + kb;
  Frag16 f;
  f.q[0] = *(const u32x4*)(p);
  f.q[1] = *(const u32x4*)(p + 16);
  return f.v;
}

__device__ inline unsigned short f32_to_bf16(float x) {
  unsigned int u = __float_as_uint(x);
  unsigned int r = u + 0x7FFFu + ((u >> 16) & 1u);   // round-to-nearest-even
  return (unsigned short)(r >> 16);
}

// ---------------------------------------------------------------- convert f32 -> bf16
__global__ __launch_bounds__(256)
void cvt_bf16_kernel(const float* __restrict__ src, unsigned short* __restrict__ dst, int n) {
  int i = blockIdx.x * blockDim.x + threadIdx.x;
  if (i < n) dst[i] = f32_to_bf16(src[i]);
}

// ---------------------------------------------------------------- GEMM: C[M,N] = A[M,K] * W[N,K]^T + bias
// wave32: 32x32 tile per wave via 4x v_wmma_f32_16x16x32_bf16 per k-step.
__global__ __launch_bounds__(256)
void gemm_bf16_kernel(const bf16_t* __restrict__ A, const bf16_t* __restrict__ W,
                      const float* __restrict__ bias, float* __restrict__ C,
                      int M, int N, int K) {
  int lane = threadIdx.x & 31;
  int wave = threadIdx.x >> 5;
  int wm = wave >> 1, wn = wave & 1;
  int m0 = blockIdx.y * 128 + wm * 32;
  int n0 = blockIdx.x * 64  + wn * 32;

  v8f c00 = {}, c01 = {}, c10 = {}, c11 = {};
  for (int k = 0; k < K; k += 32) {
    v16bf a0 = load_frag_b16(A + (size_t)m0 * K + k, K);
    v16bf a1 = load_frag_b16(A + (size_t)(m0 + 16) * K + k, K);
    v16bf b0 = load_frag_b16(W + (size_t)n0 * K + k, K);
    v16bf b1 = load_frag_b16(W + (size_t)(n0 + 16) * K + k, K);
    c00 = __builtin_amdgcn_wmma_f32_16x16x32_bf16(false, a0, false, b0, (short)0, c00, false, false);
    c01 = __builtin_amdgcn_wmma_f32_16x16x32_bf16(false, a0, false, b1, (short)0, c01, false, false);
    c10 = __builtin_amdgcn_wmma_f32_16x16x32_bf16(false, a1, false, b0, (short)0, c10, false, false);
    c11 = __builtin_amdgcn_wmma_f32_16x16x32_bf16(false, a1, false, b1, (short)0, c11, false, false);
  }
  int col0 = n0 + (lane & 15), col1 = col0 + 16;
  int rb = (lane >> 4) << 3;                 // C layout: VGPR i -> row i (lanes 0-15) / 8+i (lanes 16-31)
  float bv0 = bias[col0], bv1 = bias[col1];
#pragma unroll
  for (int i = 0; i < 8; ++i) {
    int r0 = m0 + rb + i, r1 = m0 + 16 + rb + i;
    C[(size_t)r0 * N + col0] = c00[i] + bv0;
    C[(size_t)r0 * N + col1] = c01[i] + bv1;
    C[(size_t)r1 * N + col0] = c10[i] + bv0;
    C[(size_t)r1 * N + col1] = c11[i] + bv1;
  }
}

// ---------------------------------------------------------------- hyperbolic maps
// exp_map_zero on Q/K head-vectors: write bf16 [b*h][s][64] and ||x||^2.
__global__ __launch_bounds__(256)
void qk_map_kernel(const float* __restrict__ P, unsigned short* __restrict__ out,
                   float* __restrict__ sqout) {
  int lane = threadIdx.x & 31;
  int wid  = blockIdx.x * 8 + (threadIdx.x >> 5);   // one wave per (m, h) vector
  int m = wid >> 4, h = wid & 15;
  int b = m >> 11, s = m & 2047;
  const float* v = P + (size_t)m * 1024 + h * 64;
  float v0 = v[lane], v1 = v[lane + 32];
  float ss = v0 * v0 + v1 * v1;
#pragma unroll
  for (int off = 16; off > 0; off >>= 1) ss += __shfl_xor(ss, off, 32);
  float norm = sqrtf(ss + 1e-8f);
  float t = fminf(tanhf(norm), 0.9f);
  float f = t / norm;
  float rn = sqrtf(f * f * ss + 1e-8f);
  float maxn = 1.0f - 1e-5f;
  if (rn > maxn) f *= maxn / rn;
  size_t bh = (size_t)(b * 16 + h);
  size_t base = (bh * 2048 + s) * 64;
  out[base + lane]      = f32_to_bf16(v0 * f);
  out[base + lane + 32] = f32_to_bf16(v1 * f);
  if (lane == 0) sqout[bh * 2048 + s] = f * f * ss;
}

__device__ inline void exp_then_log(float v0, float v1, float ss, float& o0, float& o1) {
  float norm = sqrtf(ss + 1e-8f);
  float t = fminf(tanhf(norm), 0.9f);
  float f = t / norm;
  float rsq = f * f * ss;
  float rn = sqrtf(rsq + 1e-8f);
  float maxn = 1.0f - 1e-5f;
  if (rn > maxn) f *= maxn / rn;
  float ysq = f * f * ss;
  float ny = fminf(sqrtf(ysq + 1e-8f), maxn);
  float arg = fminf(ny, 0.99f);
  float target = 0.5f * __logf((1.0f + arg) / (1.0f - arg));   // atanh
  float g = f * (target / ny);
  o0 = v0 * g; o1 = v1 * g;
}

// V_tan = log(exp(V)); stored TRANSPOSED bf16 [b*h][64][2048] for attn@V B-fragments.
__global__ __launch_bounds__(256)
void v_map_kernel(const float* __restrict__ P, unsigned short* __restrict__ outT) {
  int lane = threadIdx.x & 31;
  int wid  = blockIdx.x * 8 + (threadIdx.x >> 5);
  int m = wid >> 4, h = wid & 15;
  int b = m >> 11, s = m & 2047;
  const float* v = P + (size_t)m * 1024 + h * 64;
  float v0 = v[lane], v1 = v[lane + 32];
  float ss = v0 * v0 + v1 * v1;
#pragma unroll
  for (int off = 16; off > 0; off >>= 1) ss += __shfl_xor(ss, off, 32);
  float o0, o1; exp_then_log(v0, v1, ss, o0, o1);
  size_t baseT = (size_t)(b * 16 + h) * 64 * 2048;
  outT[baseT + (size_t)lane * 2048 + s]        = f32_to_bf16(o0);
  outT[baseT + (size_t)(lane + 32) * 2048 + s] = f32_to_bf16(o1);
}

// out_tan -> log(exp(out_tan)); write bf16 X[4096][1024] for final GEMM.
__global__ __launch_bounds__(256)
void post_map_kernel(const float* __restrict__ O, unsigned short* __restrict__ X) {
  int lane = threadIdx.x & 31;
  int wid  = blockIdx.x * 8 + (threadIdx.x >> 5);
  int bh = wid >> 11, s = wid & 2047;
  const float* v = O + ((size_t)bh * 2048 + s) * 64;
  float v0 = v[lane], v1 = v[lane + 32];
  float ss = v0 * v0 + v1 * v1;
#pragma unroll
  for (int off = 16; off > 0; off >>= 1) ss += __shfl_xor(ss, off, 32);
  float o0, o1; exp_then_log(v0, v1, ss, o0, o1);
  int b = bh >> 4, h = bh & 15;
  size_t m = (size_t)b * 2048 + s;
  X[m * 1024 + h * 64 + lane]      = f32_to_bf16(o0);
  X[m * 1024 + h * 64 + lane + 32] = f32_to_bf16(o1);
}

// ---------------------------------------------------------------- Poincare score from xy, ||x||^2, ||y||^2
__device__ inline float poin_score(float xy, float x2, float y2) {
  float A   = 1.0f + 2.0f * xy + y2;
  float B   = 1.0f - x2;
  float den = fmaxf(1.0f - 2.0f * xy + x2 * y2, 1e-10f);
  float mob2 = (A * A * x2 + B * B * y2 - 2.0f * A * B * xy) / (den * den);
  float nm = fminf(sqrtf(mob2 + 1e-8f), 0.95f);
  float dist = __logf((1.0f + nm) / (1.0f - nm));   // 2*atanh(nm)
  return dist * (-0.125f);                          // -dist / sqrt(64)
}

// ---------------------------------------------------------------- flash attention with hyperbolic scores
// Each wave owns 16 query rows; full online-softmax pass over 2048 keys in 32-key tiles.
__global__ __launch_bounds__(256)
void flash_kernel(const bf16_t* __restrict__ Qh, const bf16_t* __restrict__ Kh,
                  const bf16_t* __restrict__ VtT, const float* __restrict__ q2,
                  const float* __restrict__ k2, float* __restrict__ O) {
  __shared__ __align__(16) bf16_t Pst[8][16 * 32];
  int lane = threadIdx.x & 31;
  int wave = threadIdx.x >> 5;
  int bh = blockIdx.y;
  int i0 = blockIdx.x * 128 + wave * 16;

  const bf16_t* Qb  = Qh  + (size_t)bh * 2048 * 64;
  const bf16_t* Kb  = Kh  + (size_t)bh * 2048 * 64;
  const bf16_t* Vb  = VtT + (size_t)bh * 64 * 2048;
  const float*  q2b = q2  + (size_t)bh * 2048;
  const float*  k2b = k2  + (size_t)bh * 2048;

  v16bf qa0 = load_frag_b16(Qb + (size_t)i0 * 64,      64);
  v16bf qa1 = load_frag_b16(Qb + (size_t)i0 * 64 + 32, 64);

  int rb = (lane >> 4) << 3;
  float x2r[8];
#pragma unroll
  for (int i = 0; i < 8; ++i) x2r[i] = q2b[i0 + rb + i];

  float mrow[8], lrow[8];
  v8f o0 = {}, o1 = {}, o2 = {}, o3 = {};
#pragma unroll
  for (int i = 0; i < 8; ++i) { mrow[i] = -1e30f; lrow[i] = 0.0f; }

  for (int j0 = 0; j0 < 2048; j0 += 32) {
    v16bf kb00 = load_frag_b16(Kb + (size_t)j0 * 64,             64);
    v16bf kb01 = load_frag_b16(Kb + (size_t)j0 * 64 + 32,        64);
    v16bf kb10 = load_frag_b16(Kb + (size_t)(j0 + 16) * 64,      64);
    v16bf kb11 = load_frag_b16(Kb + (size_t)(j0 + 16) * 64 + 32, 64);
    v8f s0 = {}, s1 = {};
    s0 = __builtin_amdgcn_wmma_f32_16x16x32_bf16(false, qa0, false, kb00, (short)0, s0, false, false);
    s0 = __builtin_amdgcn_wmma_f32_16x16x32_bf16(false, qa1, false, kb01, (short)0, s0, false, false);
    s1 = __builtin_amdgcn_wmma_f32_16x16x32_bf16(false, qa0, false, kb10, (short)0, s1, false, false);
    s1 = __builtin_amdgcn_wmma_f32_16x16x32_bf16(false, qa1, false, kb11, (short)0, s1, false, false);

    float y20 = k2b[j0 + (lane & 15)];
    float y21 = k2b[j0 + 16 + (lane & 15)];

    float p0[8], p1[8];
#pragma unroll
    for (int i = 0; i < 8; ++i) {
      float sc0 = poin_score(s0[i], x2r[i], y20);
      float sc1 = poin_score(s1[i], x2r[i], y21);
      float vmx = fmaxf(sc0, sc1);
#pragma unroll
      for (int off = 1; off < 16; off <<= 1) vmx = fmaxf(vmx, __shfl_xor(vmx, off, 32));
      float mn = fmaxf(mrow[i], vmx);
      float scale = __expf(mrow[i] - mn);
      mrow[i] = mn;
      float e0 = __expf(sc0 - mn);
      float e1 = __expf(sc1 - mn);
      p0[i] = e0; p1[i] = e1;
      float rs = e0 + e1;
#pragma unroll
      for (int off = 1; off < 16; off <<= 1) rs += __shfl_xor(rs, off, 32);
      lrow[i] = lrow[i] * scale + rs;
      o0[i] *= scale; o1[i] *= scale; o2[i] *= scale; o3[i] *= scale;
    }
    // re-layout P (C-fragment) -> A-fragment via per-wave LDS tile
    unsigned short* ps = (unsigned short*)&Pst[wave][0];
#pragma unroll
    for (int i = 0; i < 8; ++i) {
      int row = rb + i;
      ps[row * 32 + (lane & 15)]      = f32_to_bf16(p0[i]);
      ps[row * 32 + (lane & 15) + 16] = f32_to_bf16(p1[i]);
    }
    asm volatile("s_wait_dscnt 0x0" ::: "memory");
    v16bf pa  = load_frag_b16(&Pst[wave][0], 32);
    v16bf vb0 = load_frag_b16(Vb + (size_t)0  * 2048 + j0, 2048);
    v16bf vb1 = load_frag_b16(Vb + (size_t)16 * 2048 + j0, 2048);
    v16bf vb2 = load_frag_b16(Vb + (size_t)32 * 2048 + j0, 2048);
    v16bf vb3 = load_frag_b16(Vb + (size_t)48 * 2048 + j0, 2048);
    o0 = __builtin_amdgcn_wmma_f32_16x16x32_bf16(false, pa, false, vb0, (short)0, o0, false, false);
    o1 = __builtin_amdgcn_wmma_f32_16x16x32_bf16(false, pa, false, vb1, (short)0, o1, false, false);
    o2 = __builtin_amdgcn_wmma_f32_16x16x32_bf16(false, pa, false, vb2, (short)0, o2, false, false);
    o3 = __builtin_amdgcn_wmma_f32_16x16x32_bf16(false, pa, false, vb3, (short)0, o3, false, false);
    asm volatile("s_wait_dscnt 0x0" ::: "memory");   // LDS reads done before next tile overwrites
  }

  float* Ob = O + (size_t)bh * 2048 * 64;
  int col = lane & 15;
#pragma unroll
  for (int i = 0; i < 8; ++i) {
    int row = i0 + rb + i;
    float inv = 1.0f / lrow[i];
    Ob[(size_t)row * 64 + col]      = o0[i] * inv;
    Ob[(size_t)row * 64 + col + 16] = o1[i] * inv;
    Ob[(size_t)row * 64 + col + 32] = o2[i] * inv;
    Ob[(size_t)row * 64 + col + 48] = o3[i] * inv;
  }
}

// ---------------------------------------------------------------- host launcher
extern "C" void kernel_launch(void* const* d_in, const int* in_sizes, int n_in,
                              void* d_out, int out_size, void* d_ws, size_t ws_size,
                              hipStream_t stream) {
  const float* H  = (const float*)d_in[0];
  const float* Wq = (const float*)d_in[1];
  const float* bq = (const float*)d_in[2];
  const float* Wk = (const float*)d_in[3];
  const float* bk = (const float*)d_in[4];
  const float* Wv = (const float*)d_in[5];
  const float* bv = (const float*)d_in[6];
  const float* Wo = (const float*)d_in[7];
  const float* bo = (const float*)d_in[8];
  float* out = (float*)d_out;

  const size_t MB = 1u << 20;
  char* ws = (char*)d_ws;
  unsigned short* Hbf  = (unsigned short*)(ws);              //  8 MB: H bf16 [4096][1024]
  unsigned short* Wqbf = (unsigned short*)(ws + 8  * MB);    //  2 MB each
  unsigned short* Wkbf = (unsigned short*)(ws + 10 * MB);
  unsigned short* Wvbf = (unsigned short*)(ws + 12 * MB);
  unsigned short* Wobf = (unsigned short*)(ws + 14 * MB);
  float*          Praw = (float*)(ws + 16 * MB);             // 16 MB: raw proj (reused 3x, then O)
  unsigned short* Qh   = (unsigned short*)(ws + 32 * MB);    //  8 MB: Q_hyp bf16 [32][2048][64]
  unsigned short* Kh   = (unsigned short*)(ws + 40 * MB);    //  8 MB
  unsigned short* VtT  = (unsigned short*)(ws + 48 * MB);    //  8 MB: V_tan^T bf16 [32][64][2048]
  float*          q2   = (float*)(ws + 56 * MB);             // 256 KB
  float*          k2   = (float*)(ws + 56 * MB + 256 * 1024);
  unsigned short* Xbf  = (unsigned short*)(ws + 57 * MB);    //  8 MB

  const int MKH = 4096 * 1024;   // H elements
  const int MKW = 1024 * 1024;   // W elements

  cvt_bf16_kernel<<<MKH / 256, 256, 0, stream>>>(H,  Hbf,  MKH);
  cvt_bf16_kernel<<<MKW / 256, 256, 0, stream>>>(Wq, Wqbf, MKW);
  cvt_bf16_kernel<<<MKW / 256, 256, 0, stream>>>(Wk, Wkbf, MKW);
  cvt_bf16_kernel<<<MKW / 256, 256, 0, stream>>>(Wv, Wvbf, MKW);
  cvt_bf16_kernel<<<MKW / 256, 256, 0, stream>>>(Wo, Wobf, MKW);

  dim3 ggrid(1024 / 64, 4096 / 128);
  dim3 mgrid(65536 / 8);          // 65536 head-vectors, 8 waves per block

  // Q projection + exp_map
  gemm_bf16_kernel<<<ggrid, 256, 0, stream>>>((const bf16_t*)Hbf, (const bf16_t*)Wqbf, bq, Praw, 4096, 1024, 1024);
  qk_map_kernel<<<mgrid, 256, 0, stream>>>(Praw, Qh, q2);
  // K projection + exp_map
  gemm_bf16_kernel<<<ggrid, 256, 0, stream>>>((const bf16_t*)Hbf, (const bf16_t*)Wkbf, bk, Praw, 4096, 1024, 1024);
  qk_map_kernel<<<mgrid, 256, 0, stream>>>(Praw, Kh, k2);
  // V projection + log(exp(V)) transposed
  gemm_bf16_kernel<<<ggrid, 256, 0, stream>>>((const bf16_t*)Hbf, (const bf16_t*)Wvbf, bv, Praw, 4096, 1024, 1024);
  v_map_kernel<<<mgrid, 256, 0, stream>>>(Praw, VtT);

  // flash attention (O reuses Praw: [32][2048][64] f32)
  dim3 fgrid(2048 / 128, 32);
  flash_kernel<<<fgrid, 256, 0, stream>>>((const bf16_t*)Qh, (const bf16_t*)Kh, (const bf16_t*)VtT,
                                          q2, k2, Praw);

  // out map + final projection
  post_map_kernel<<<mgrid, 256, 0, stream>>>(Praw, Xbf);
  gemm_bf16_kernel<<<ggrid, 256, 0, stream>>>((const bf16_t*)Xbf, (const bf16_t*)Wobf, bo, out, 4096, 1024, 1024);
}
